// Foil_32719060861633
// MI455X (gfx1250) — matrix operand order repaired and checked by reading, not compiled, not run on hardware
//
#include <hip/hip_runtime.h>

#define GROUPS 4
#define POINTS 120
#define TAB_STRIDE 364   // [0]=dmin [1]=width [2..121]=accum [122..241]=ftheta [242..361]=fvelo
#define WS_MINMAX 0      // 8 uints: {min_enc,max_enc} per group
#define WS_HIST   8      // 4*120 uints
#define WS_TAB    488    // 4*364 floats

typedef __attribute__((ext_vector_type(16))) _Float16 v16h;
typedef __attribute__((ext_vector_type(8)))  float    v8f;
typedef __attribute__((ext_vector_type(4)))  float    v4f;

// Map float -> order-preserving uint (for uint atomicMin/Max), and back.
__device__ __forceinline__ unsigned fenc(float f) {
    unsigned u = __float_as_uint(f);
    return (u & 0x80000000u) ? ~u : (u | 0x80000000u);
}
__device__ __forceinline__ float fdec(unsigned u) {
    return __uint_as_float((u & 0x80000000u) ? (u & 0x7FFFFFFFu) : ~u);
}

// j -> input index for group ix:  j = ((b*16+c_sub)*256+h)*256+w,
// input = ((b*64 + 4*c_sub + ix)*256+h)*256+w = ((4*(j>>16)+ix)<<16) | (j&0xFFFF)
__device__ __forceinline__ unsigned in_index(unsigned j, int ix) {
    return (((j >> 16) * 4u + (unsigned)ix) << 16) | (j & 0xFFFFu);
}

__global__ void foil_init(unsigned* gmm, unsigned* ghist) {
    int i = threadIdx.x;
    if (i < 2 * GROUPS) gmm[i] = (i & 1) ? 0u : 0xFFFFFFFFu;
    for (int k = i; k < GROUPS * POINTS; k += blockDim.x) ghist[k] = 0u;
}

__global__ __launch_bounds__(256) void foil_minmax(const float* __restrict__ in,
                                                   const float* __restrict__ ctp,
                                                   unsigned* __restrict__ gmm, int nj) {
    __shared__ unsigned smn[GROUPS], smx[GROUPS];
    int tid = threadIdx.x;
    if (tid < GROUPS) { smn[tid] = 0xFFFFFFFFu; smx[tid] = 0u; }
    __syncthreads();
    float ct = *ctp;
    unsigned mn[GROUPS], mx[GROUPS];
#pragma unroll
    for (int ix = 0; ix < GROUPS; ++ix) { mn[ix] = 0xFFFFFFFFu; mx[ix] = 0u; }
    unsigned stride = (unsigned)gridDim.x * blockDim.x;
    for (unsigned j = blockIdx.x * blockDim.x + tid; j < (unsigned)nj; j += stride) {
#pragma unroll
        for (int ix = 0; ix < GROUPS; ++ix) {
            unsigned e = fenc(in[in_index(j, ix)] * ct);
            mn[ix] = min(mn[ix], e);
            mx[ix] = max(mx[ix], e);
        }
    }
#pragma unroll
    for (int ix = 0; ix < GROUPS; ++ix) {
        for (int off = 16; off > 0; off >>= 1) {   // wave32 reduction
            mn[ix] = min(mn[ix], (unsigned)__shfl_xor((int)mn[ix], off, 32));
            mx[ix] = max(mx[ix], (unsigned)__shfl_xor((int)mx[ix], off, 32));
        }
    }
    if ((tid & 31) == 0) {
#pragma unroll
        for (int ix = 0; ix < GROUPS; ++ix) {
            atomicMin(&smn[ix], mn[ix]);
            atomicMax(&smx[ix], mx[ix]);
        }
    }
    __syncthreads();
    if (tid < GROUPS) {
        atomicMin(&gmm[2 * tid + 0], smn[tid]);
        atomicMax(&gmm[2 * tid + 1], smx[tid]);
    }
}

__global__ __launch_bounds__(256) void foil_hist(const float* __restrict__ in,
                                                 const float* __restrict__ ctp,
                                                 const unsigned* __restrict__ gmm,
                                                 unsigned* __restrict__ ghist, int nj) {
    __shared__ unsigned sh[GROUPS * POINTS];
    __shared__ float sdmin[GROUPS], sinvw[GROUPS];
    int tid = threadIdx.x;
    for (int i = tid; i < GROUPS * POINTS; i += blockDim.x) sh[i] = 0u;
    if (tid < GROUPS) {
        float dmn = fdec(gmm[2 * tid + 0]);
        float dmx = fdec(gmm[2 * tid + 1]);
        sdmin[tid] = dmn;
        sinvw[tid] = (float)POINTS / (dmx - dmn);
    }
    __syncthreads();
    float ct = *ctp;
    unsigned stride = (unsigned)gridDim.x * blockDim.x;
    for (unsigned j = blockIdx.x * blockDim.x + tid; j < (unsigned)nj; j += stride) {
#pragma unroll
        for (int ix = 0; ix < GROUPS; ++ix) {
            float d = in[in_index(j, ix)] * ct;
            int b = (int)floorf((d - sdmin[ix]) * sinvw[ix]);
            b = min(max(b, 0), POINTS - 1);
            atomicAdd(&sh[ix * POINTS + b], 1u);
        }
    }
    __syncthreads();
    for (int i = tid; i < GROUPS * POINTS; i += blockDim.x)
        if (sh[i]) atomicAdd(&ghist[i], sh[i]);
}

// One block, 4 waves; wave g owns group g. CDF cumsum done as a triangular
// matmul on the WMMA unit: pad 120->128 = 8 chunks of 16, R = L16 x V with a
// single v_wmma_f32_16x16x32_f16 (L16 is 0/1 -> exact in f16), then an 8-wide
// shuffle scan of the chunk totals. Then warp theta/velo through the inverse CDF.
__global__ __launch_bounds__(128) void foil_tables(const unsigned* __restrict__ gmm,
                                                   const unsigned* __restrict__ ghist,
                                                   const float* __restrict__ params,
                                                   float* __restrict__ tab, float scale) {
    __shared__ float sp[GROUPS][256];    // padded probs (x (P-1)/N), zeros beyond 120
    __shared__ float sacc[GROUPS][128];  // scaled CDF
    const int tid  = threadIdx.x;
    const int g    = tid >> 5;
    const int lane = tid & 31;
    const int l15  = lane & 15;
    const int hig  = lane >> 4;

    for (int i = lane; i < 256; i += 32)
        sp[g][i] = (i < POINTS) ? (float)ghist[g * POINTS + i] * scale : 0.0f;
    __syncthreads();

    // A = L16 in 16x32 f16 A-layout: lanes 0-15 M=l15 hold K=(0..7,16..23),
    // lanes 16-31 hold K=(8..15,24..31). Columns K>=16 are zero padding.
    // B = V (32x16): lanes 0-15 hold K=(0..7,16..23), lanes 16-31 K=(8..15,24..31) of column N=l15; K>=16 zero.
    v16h A, Bv;
#pragma unroll
    for (int e = 0; e < 8; ++e) {
        int K = (hig ? 8 : 0) + e;
        A[e]     = (_Float16)((K <= l15) ? 1.0f : 0.0f);
        A[e + 8] = (_Float16)0.0f;
        Bv[e]     = (_Float16)sp[g][l15 * 16 + (hig ? 8 : 0) + e];
        Bv[e + 8] = (_Float16)0.0f;
    }
    v8f Cz = {};
    v8f D = __builtin_amdgcn_wmma_f32_16x16x32_f16(false, A, false, Bv,
                                                   (short)0, Cz, false, false);
    // D: VGPR r, lanes 0-15 -> (M=r, N=lane); lanes 16-31 -> (M=8+r, N=lane-16).
    // R[m][n] = inclusive scan of chunk n up to m. Chunk total = R[15][n] (lane 16+n, r=7).
    float d7 = D[7];
    float off = 0.0f;
#pragma unroll
    for (int t = 0; t < 8; ++t) {
        float tot = __shfl(d7, 16 + t, 32);
        if (t < l15) off += tot;
    }
#pragma unroll
    for (int r = 0; r < 8; ++r) {
        int idx = l15 * 16 + (hig ? 8 : 0) + r;
        if (idx < POINTS) sacc[g][idx] = D[r] + off;
    }
    __syncthreads();

    float dmn = fdec(gmm[2 * g + 0]);
    float dmx = fdec(gmm[2 * g + 1]);
    float width = (dmx - dmn) * (1.0f / POINTS);
    float* T = tab + g * TAB_STRIDE;
    if (lane == 0) { T[0] = dmn; T[1] = width; }
    for (int i = lane; i < POINTS; i += 32) T[2 + i] = sacc[g][i];

    const float* thp = params + (0 * GROUPS + g) * POINTS;
    const float* vep = params + (1 * GROUPS + g) * POINTS;
    for (int k = lane; k < POINTS; k += 32) {
        float xk = (float)k;
        int lo = 0, hi2 = POINTS;                 // searchsorted(accum, k, 'left')
        while (lo < hi2) {
            int mid = (lo + hi2) >> 1;
            if (sacc[g][mid] < xk) lo = mid + 1; else hi2 = mid;
        }
        int ind = min(max(lo - 1, 0), POINTS - 2);
        float x0 = sacc[g][ind];
        float t = (xk - x0) / (sacc[g][ind + 1] - x0);
        T[2 + POINTS + k]     = thp[ind] + t * (thp[ind + 1] - thp[ind]);
        T[2 + 2 * POINTS + k] = vep[ind] + t * (vep[ind + 1] - vep[ind]);
    }
}

__global__ __launch_bounds__(256) void foil_apply(const float* __restrict__ in,
                                                  const float* __restrict__ ctp,
                                                  const float* __restrict__ stp,
                                                  const float* __restrict__ tab,
                                                  float* __restrict__ out, int nj) {
    __shared__ float stab[GROUPS * TAB_STRIDE];
    for (int i = threadIdx.x; i < GROUPS * TAB_STRIDE; i += blockDim.x) stab[i] = tab[i];
    __syncthreads();
    float ct = *ctp, sc = *stp;
    unsigned j = blockIdx.x * 256u + threadIdx.x;
    if (j >= (unsigned)nj) return;
    float r[GROUPS];
#pragma unroll
    for (int ix = 0; ix < GROUPS; ++ix) {
        float d = in[in_index(j, ix)] * ct;
        const float* T = &stab[ix * TAB_STRIDE];
        float dmin = T[0], w = T[1], invw = 1.0f / w;
        // searchsorted on the UNIFORM bin-center grid collapses to a floor.
        float u = (d - dmin) * invw - 0.5f;
        int ind = min(max((int)floorf(u), 0), POINTS - 2);
        float g0 = dmin + w * ((float)ind + 0.5f);
        float t = (d - g0) * invw;
        float a0 = T[2 + ind];
        float index = a0 + t * (T[3 + ind] - a0);
        int bg = min(max((int)floorf(index), 0), POINTS - 1);
        float pos = index - (float)bg;
        int en = min(bg + 1, POINTS - 1);
        float th = (1.0f - pos) * T[2 + POINTS + bg] + pos * T[2 + POINTS + en];
        float ve = (1.0f - pos) * T[2 + 2 * POINTS + bg] + pos * T[2 + 2 * POINTS + en];
        float ds = ve * 0.01f;
        r[ix] = (d * (1.0f + ds * __sinf(th)) + ds * __cosf(th)) * sc;
    }
    v4f res = { r[0], r[1], r[2], r[3] };          // out_flat = 4*j + ix -> one float4
    __builtin_nontemporal_store(res, (v4f*)out + j); // NT: don't evict cached input from L2
}

extern "C" void kernel_launch(void* const* d_in, const int* in_sizes, int n_in,
                              void* d_out, int out_size, void* d_ws, size_t ws_size,
                              hipStream_t stream) {
    const float* data   = (const float*)d_in[0];
    const float* params = (const float*)d_in[1];
    const float* ct     = (const float*)d_in[2];
    const float* st     = (const float*)d_in[3];
    float*    out  = (float*)d_out;
    unsigned* ws   = (unsigned*)d_ws;
    unsigned* gmm  = ws + WS_MINMAX;
    unsigned* ghist= ws + WS_HIST;
    float*    tab  = (float*)(ws + WS_TAB);
    int nj = in_sizes[0] / GROUPS;   // elements per group = 8,388,608

    foil_init<<<1, 512, 0, stream>>>(gmm, ghist);
    foil_minmax<<<1024, 256, 0, stream>>>(data, ct, gmm, nj);
    foil_hist<<<1024, 256, 0, stream>>>(data, ct, gmm, ghist, nj);
    foil_tables<<<1, 128, 0, stream>>>(gmm, ghist, params, tab,
                                       (float)(POINTS - 1) / (float)nj);
    foil_apply<<<(nj + 255) / 256, 256, 0, stream>>>(data, ct, st, tab, out, nj);
}
